// AIGEncoder_24163486007361
// MI455X (gfx1250) — matrix operand order repaired and checked by reading, not compiled, not run on hardware
//
#include <hip/hip_runtime.h>
#include <stdint.h>

#define N_NODES 50000
#define N_EDGES 320000
#define DIM     256
#define N_GRAPHS 64
#define BM 128
#define BN 128

typedef __attribute__((ext_vector_type(16))) __bf16 v16bf;
typedef __attribute__((ext_vector_type(8)))  __bf16 v8bf;
typedef __attribute__((ext_vector_type(8)))  float  v8f;

__device__ __forceinline__ unsigned short f2bf(float f) {
  union { float f; unsigned u; } v; v.f = f;
  unsigned r = v.u + 0x7FFFu + ((v.u >> 16) & 1u);
  return (unsigned short)(r >> 16);
}

__device__ __forceinline__ void atomAddF(float* p, float v) {
  __hip_atomic_fetch_add(p, v, __ATOMIC_RELAXED, __HIP_MEMORY_SCOPE_AGENT);
}

// ---- weight convert + transpose: W[k][n] (f32 row-major) -> Wt[n][k] (bf16) ----
__global__ void k_wt(const float* __restrict__ W, unsigned short* __restrict__ Wt) {
  int idx = blockIdx.x * blockDim.x + threadIdx.x;   // 65536 threads
  int k = idx >> 8, n = idx & 255;
  Wt[n * DIM + k] = f2bf(W[k * DIM + n]);
}

// ---- vectorized f32 copy ----
__global__ void k_copy4(const float4* __restrict__ src, float4* __restrict__ dst, int n4) {
  int i = blockIdx.x * blockDim.x + threadIdx.x;
  if (i < n4) dst[i] = src[i];
}

// ---- edge message + scatter-add: agg[dst] += relu(h[src] + ea) ----
// 256 threads / block, 4 edges / block, 64 lanes x float4 cover one 256-f32 row
__global__ void k_scatter(const float* __restrict__ h, const long long* __restrict__ ei,
                          const float* __restrict__ ea, float* __restrict__ agg) {
  int e = blockIdx.x * 4 + (threadIdx.x >> 6);
  int q = threadIdx.x & 63;
  long long s = ei[e];
  long long t = ei[N_EDGES + e];
  float4 hv = ((const float4*)(h + (size_t)s * DIM))[q];
  float4 ev = ((const float4*)(ea + (size_t)e * DIM))[q];
  float m0 = fmaxf(hv.x + ev.x, 0.0f);
  float m1 = fmaxf(hv.y + ev.y, 0.0f);
  float m2 = fmaxf(hv.z + ev.z, 0.0f);
  float m3 = fmaxf(hv.w + ev.w, 0.0f);
  float* p = agg + (size_t)t * DIM + q * 4;
  atomAddF(p + 0, m0); atomAddF(p + 1, m1);
  atomAddF(p + 2, m2); atomAddF(p + 3, m3);
}

// ---- f32 -> bf16 (RNE), 4 elems / thread ----
__global__ void k_f2bf(const float4* __restrict__ src, uint2* __restrict__ dst, int n4) {
  int i = blockIdx.x * blockDim.x + threadIdx.x;
  if (i < n4) {
    float4 v = src[i];
    uint2 o;
    o.x = (unsigned)f2bf(v.x) | ((unsigned)f2bf(v.y) << 16);
    o.y = (unsigned)f2bf(v.z) | ((unsigned)f2bf(v.w) << 16);
    dst[i] = o;
  }
}

// ---- WMMA GEMM: out = relu(A[M x 256](bf16) @ W(via Wt col-major bf16) + bias) ----
// 8 waves; wave w -> rows [blk*128 + 16w, +16), 8 column tiles of 16 over BN=128.
// All fragment loads are single-base + compile-time immediate offsets.
__global__ void __launch_bounds__(256)
k_gemm(const unsigned short* __restrict__ A, const unsigned short* __restrict__ Wt,
       const float* __restrict__ bias, float* __restrict__ outF,
       unsigned short* __restrict__ outB, int M, int store_bf16) {
  int wave = threadIdx.x >> 5;
  int lane = threadIdx.x & 31;
  int r16  = lane & 15;
  int half = lane >> 4;
  int rowBase = blockIdx.x * BM + wave * 16;
  int colBase = blockIdx.y * BN;

  // A base: row = rowBase + r16 (lane=M), with the 8*half K-skew folded in.
  const unsigned short* Ap = A + (size_t)(rowBase + r16) * DIM + 8 * half;
  // B base: column n = colBase + r16 (lane=N), 8*half K-skew folded in.
  const unsigned short* Bp = Wt + (size_t)(colBase + r16) * DIM + 8 * half;

  v8f acc[8] = {};

#pragma unroll
  for (int k0 = 0; k0 < DIM; k0 += 32) {
    // A fragment (ISA 16-bit A layout): elems 0..7 = K(k0+8h..), elems 8..15 = K(k0+16+8h..)
    v8bf alo = *(const v8bf*)(Ap + k0);
    v8bf ahi = *(const v8bf*)(Ap + k0 + 16);
    v16bf af;
#pragma unroll
    for (int i = 0; i < 8; ++i) { af[i] = alo[i]; af[i + 8] = ahi[i]; }

#pragma unroll
    for (int t = 0; t < 8; ++t) {
      // B fragment: same K layout as A (elems 0..7 = K(k0+8h..), 8..15 = K(k0+16+8h..))
      v8bf blo = *(const v8bf*)(Bp + t * 16 * DIM + k0);
      v8bf bhi = *(const v8bf*)(Bp + t * 16 * DIM + k0 + 16);
      v16bf bf;
#pragma unroll
      for (int i = 0; i < 8; ++i) { bf[i] = blo[i]; bf[i + 8] = bhi[i]; }
      acc[t] = __builtin_amdgcn_wmma_f32_16x16x32_bf16(
          false, af, false, bf, (short)0, acc[t], false, false);
    }
  }

  // Epilogue: C/D layout lanes0-15: N=lane, M=r; lanes16-31: N=lane-16, M=8+r
  const float* bp = bias + colBase + r16;
  int rbase = rowBase + 8 * half;            // this lane's first output row
  if (rowBase + BM <= M) {
    // full tile: unguarded stores, base + immediate offsets
    if (store_bf16) {
      unsigned short* ob = outB + (size_t)rbase * DIM + colBase + r16;
#pragma unroll
      for (int t = 0; t < 8; ++t) {
        float bv = bp[t * 16];
#pragma unroll
        for (int r = 0; r < 8; ++r)
          ob[(size_t)r * DIM + t * 16] = f2bf(fmaxf(acc[t][r] + bv, 0.0f));
      }
    } else {
      float* of = outF + (size_t)rbase * DIM + colBase + r16;
#pragma unroll
      for (int t = 0; t < 8; ++t) {
        float bv = bp[t * 16];
#pragma unroll
        for (int r = 0; r < 8; ++r)
          of[(size_t)r * DIM + t * 16] = fmaxf(acc[t][r] + bv, 0.0f);
      }
    }
  } else {
    // tail row-block: per-row guard
#pragma unroll
    for (int t = 0; t < 8; ++t) {
      int col = colBase + t * 16 + r16;
      float bv = bp[t * 16];
#pragma unroll
      for (int r = 0; r < 8; ++r) {
        int row = rbase + r;
        if (row < M) {
          float v = fmaxf(acc[t][r] + bv, 0.0f);
          if (store_bf16) outB[(size_t)row * DIM + col] = f2bf(v);
          else            outF[(size_t)row * DIM + col] = v;
        }
      }
    }
  }
}

// ---- pooling ----
__global__ void k_zero(float* __restrict__ p, int n) {
  int i = blockIdx.x * blockDim.x + threadIdx.x;
  if (i < n) p[i] = 0.0f;
}

__global__ void k_pool(const float* __restrict__ h, const long long* __restrict__ batch,
                       float* __restrict__ sums, float* __restrict__ cnt) {
  int node = blockIdx.x;
  int q = threadIdx.x;                     // 64 lanes x float4 = full row
  long long g = batch[node];
  float4 v = ((const float4*)(h + (size_t)node * DIM))[q];
  float* p = sums + (size_t)g * DIM + q * 4;
  atomAddF(p + 0, v.x); atomAddF(p + 1, v.y);
  atomAddF(p + 2, v.z); atomAddF(p + 3, v.w);
  if (q == 0) atomAddF(cnt + g, 1.0f);
}

__global__ void k_final(const float* __restrict__ sums, const float* __restrict__ cnt,
                        float* __restrict__ out) {
  int i = blockIdx.x * blockDim.x + threadIdx.x;   // G*DIM
  int g = i >> 8;
  out[i] = sums[i] / fmaxf(cnt[g], 1.0f);
}

extern "C" void kernel_launch(void* const* d_in, const int* in_sizes, int n_in,
                              void* d_out, int out_size, void* d_ws, size_t ws_size,
                              hipStream_t stream) {
  const float*      x   = (const float*)d_in[0];
  const long long*  ei  = (const long long*)d_in[1];
  const float*      ea  = (const float*)d_in[2];
  const long long*  bat = (const long long*)d_in[3];
  const float*      W1  = (const float*)d_in[4];
  const float*      b1  = (const float*)d_in[5];
  const float*      W2  = (const float*)d_in[6];
  const float*      b2  = (const float*)d_in[7];
  float* out = (float*)d_out;

  // workspace carve-up (rows padded by 128 so tail-tile A loads stay in-bounds)
  const size_t NP = N_NODES + 128;
  char* w = (char*)d_ws;
  float*          agg  = (float*)w;           w += NP * DIM * sizeof(float);
  float*          hbuf = (float*)w;           w += NP * DIM * sizeof(float);
  unsigned short* abf  = (unsigned short*)w;  w += NP * DIM * sizeof(unsigned short);
  unsigned short* zbf  = (unsigned short*)w;  w += NP * DIM * sizeof(unsigned short);
  unsigned short* wt1  = (unsigned short*)w;  w += DIM * DIM * sizeof(unsigned short);
  unsigned short* wt2  = (unsigned short*)w;  w += DIM * DIM * sizeof(unsigned short);
  float*          sums = (float*)w;           w += N_GRAPHS * DIM * sizeof(float);
  float*          cnt  = (float*)w;           w += N_GRAPHS * sizeof(float);

  const int n4   = N_NODES * DIM / 4;          // 3.2M float4
  const int cpyG = (n4 + 255) / 256;
  dim3 gemmGrid((N_NODES + BM - 1) / BM, DIM / BN);  // (391, 2)

  // weights -> bf16, transposed
  k_wt<<<256, 256, 0, stream>>>(W1, wt1);
  k_wt<<<256, 256, 0, stream>>>(W2, wt2);

  // ---------- layer 1 ----------
  k_copy4 <<<cpyG, 256, 0, stream>>>((const float4*)x, (float4*)agg, n4);
  k_scatter<<<N_EDGES / 4, 256, 0, stream>>>(x, ei, ea, agg);
  k_f2bf  <<<cpyG, 256, 0, stream>>>((const float4*)agg, (uint2*)abf, n4);
  k_gemm  <<<gemmGrid, 256, 0, stream>>>(abf, wt1, b1, nullptr, zbf, N_NODES, 1);
  k_gemm  <<<gemmGrid, 256, 0, stream>>>(zbf, wt2, b2, hbuf, nullptr, N_NODES, 0);

  // ---------- layer 2 ----------
  k_copy4 <<<cpyG, 256, 0, stream>>>((const float4*)hbuf, (float4*)agg, n4);
  k_scatter<<<N_EDGES / 4, 256, 0, stream>>>(hbuf, ei, ea, agg);
  k_f2bf  <<<cpyG, 256, 0, stream>>>((const float4*)agg, (uint2*)abf, n4);
  k_gemm  <<<gemmGrid, 256, 0, stream>>>(abf, wt1, b1, nullptr, zbf, N_NODES, 1);
  k_gemm  <<<gemmGrid, 256, 0, stream>>>(zbf, wt2, b2, agg /*h2 reuse*/, nullptr, N_NODES, 0);

  // ---------- global mean pool ----------
  k_zero <<<(N_GRAPHS * DIM + N_GRAPHS + 255) / 256, 256, 0, stream>>>(sums, N_GRAPHS * DIM + N_GRAPHS);
  k_pool <<<N_NODES, 64, 0, stream>>>(agg, bat, sums, cnt);
  k_final<<<N_GRAPHS, 256, 0, stream>>>(sums, cnt, out);
}
